// MultiHeadAttention_20349555048776
// MI455X (gfx1250) — compile-verified, matrix operand order checked by reading, not run on hardware
//
#include <hip/hip_runtime.h>
#include <hip/hip_bf16.h>
#include <stdint.h>

// ---------------------------------------------------------------------------
// MHA forward for MI455X (gfx1250, wave32, WMMA).
//   x(2,2048,1024) @ Wqkv -> qkv ; causal flash attention ; @ Wout -> out
// Precision: bf16 operands, f32 accumulation (v_wmma_f32_16x16x32_bf16).
// ~69 GFLOP vs ~85 MB traffic @ 23.3 TB/s -> matrix-core bound; fits in L2.
// Data movement: weights pre-transposed once so GEMM A and B tiles both stage
// via GLOBAL_LOAD_ASYNC_TO_LDS_B128 (ASYNCcnt) with double-buffered LDS.
// ---------------------------------------------------------------------------

typedef __attribute__((ext_vector_type(16))) __bf16 v16bf;
typedef __attribute__((ext_vector_type(8)))  float  v8f;
typedef unsigned short u16;

#define Bc   2
#define S_   2048
#define D_   1024
#define H_   16
#define HD_  64
#define BS_  4096          // B*S
#define BH_  32            // B*H
#define SCALE_ 0.125f      // HD^-0.5

union Frag { uint4 q[2]; v16bf v; };

__device__ inline u16 f2bf(float f) {
  union { float f; unsigned u; } v; v.f = f;
  unsigned u = v.u;
  unsigned r = u + 0x7FFFu + ((u >> 16) & 1u);        // round-to-nearest-even
  if ((u & 0x7F800000u) == 0x7F800000u) r = u;        // Inf/NaN passthrough
  return (u16)(r >> 16);
}

__device__ inline v8f wmma_bf16(v16bf a, v16bf b, v8f c) {
  // (neg_a, A, neg_b, B, c_mod, C, reuse_a, reuse_b)
  return __builtin_amdgcn_wmma_f32_16x16x32_bf16(false, a, false, b, (short)0, c,
                                                 false, false);
}

// CDNA5 async global->LDS copy, 16B per lane, tracked by ASYNCcnt.
// LDS address = low 32 bits of the generic shared pointer (aperture in MSBs).
__device__ inline void async_g2l_b128(void* lds, const void* g) {
  unsigned loff = (unsigned)(uintptr_t)lds;
  asm volatile("global_load_async_to_lds_b128 %0, %1, off"
               :: "v"(loff), "v"(g) : "memory");
}
#define WAIT_ASYNC(n) asm volatile("s_wait_asynccnt " #n ::: "memory")

// ---------------------------------------------------------------------------
// fp32 -> bf16 elementwise convert (for x)
// ---------------------------------------------------------------------------
__global__ void f32_to_bf16(const float* __restrict__ src, u16* __restrict__ dst,
                            int n) {
  int i = blockIdx.x * blockDim.x + threadIdx.x;
  if (i < n) dst[i] = f2bf(src[i]);
}

// ---------------------------------------------------------------------------
// One-shot weight prep: W (KxN f32, row-major) -> Wt (NxK bf16, row-major)
// 32x32 LDS tile transpose, coalesced on both sides.
// ---------------------------------------------------------------------------
__global__ __launch_bounds__(256) void transpose_f32_to_bf16(
    const float* __restrict__ W, u16* __restrict__ Wt, int K, int N)
{
  __shared__ u16 tile[32][33];
  const int k0 = blockIdx.y * 32, n0 = blockIdx.x * 32;
  const int tx = threadIdx.x & 31, ty = threadIdx.x >> 5;   // 32 x 8
#pragma unroll
  for (int r = 0; r < 32; r += 8)
    tile[ty + r][tx] = f2bf(W[(size_t)(k0 + ty + r) * N + n0 + tx]);
  __syncthreads();
#pragma unroll
  for (int r = 0; r < 32; r += 8)
    Wt[(size_t)(n0 + ty + r) * K + k0 + tx] = tile[tx][ty + r];
}

// ---------------------------------------------------------------------------
// bf16 WMMA GEMM: C(MxN) = A(MxK) * Bt(NxK)^T + bias(N)
// Block tile 128x128, 8 waves, wave tile 64x32 (4x2 frags), K-step 32.
// A and B tiles staged with async global->LDS copies, double buffered:
//   issue tile i+1 (4 async ops/wave) ; s_wait_asynccnt 4 ; barrier ; compute.
// mode 0: scatter qkv -> Q/K (b,h,s,hd) and V transposed (b,h,hd,s), bf16
// mode 1: write f32 row-major output
// ---------------------------------------------------------------------------
__global__ __launch_bounds__(256) void gemm_bf16(
    const u16* __restrict__ A, const u16* __restrict__ Bt,
    const float* __restrict__ bias, int M, int N, int K, int mode,
    u16* __restrict__ Qb, u16* __restrict__ Kb, u16* __restrict__ Vt,
    float* __restrict__ outF)
{
  constexpr int ST = 40;                               // padded LDS stride
  __shared__ __align__(16) u16 ldsA[2][128 * ST];      // 128 x 32 bf16, x2
  __shared__ __align__(16) u16 ldsB[2][128 * ST];      // 128(n) x 32(k), x2

  const int t    = threadIdx.x;
  const int lane = t & 31;
  const int wid  = t >> 5;
  const int l16  = lane & 15;
  const int lh   = lane >> 4;
  const int m0   = blockIdx.y * 128;
  const int n0   = blockIdx.x * 128;
  const int waveM = (wid & 1) * 64;
  const int waveN = (wid >> 1) * 32;

  v8f acc[4][2];
#pragma unroll
  for (int i = 0; i < 4; ++i)
#pragma unroll
    for (int j = 0; j < 2; ++j)
      acc[i][j] = (v8f){0.f,0.f,0.f,0.f,0.f,0.f,0.f,0.f};

  // 512 16B chunks per 128x32 tile -> 2 chunks per thread per matrix.
  auto issue_tile = [&](int buf, int kk) {
#pragma unroll
    for (int i = 0; i < 2; ++i) {
      int c = t + i * 256, row = c >> 2, seg = c & 3;
      async_g2l_b128(&ldsA[buf][row * ST + seg * 8],
                     A  + (size_t)(m0 + row) * K + kk + seg * 8);
      async_g2l_b128(&ldsB[buf][row * ST + seg * 8],
                     Bt + (size_t)(n0 + row) * K + kk + seg * 8);
    }
  };

  issue_tile(0, 0);
  const int T = K >> 5;
  for (int it = 0; it < T; ++it) {
    const int cur = it & 1;
    if (it + 1 < T) {
      issue_tile(1 - cur, (it + 1) << 5);
      WAIT_ASYNC(0x4);                 // tile `it` done (in-order), next in flight
    } else {
      WAIT_ASYNC(0x0);
    }
    __syncthreads();                   // all waves' tile data visible

    Frag bfr[2];
#pragma unroll
    for (int fn = 0; fn < 2; ++fn) {
      const u16* p = &ldsB[cur][(waveN + fn * 16 + l16) * ST + lh * 8];
      bfr[fn].q[0] = *(const uint4*)(p);
      bfr[fn].q[1] = *(const uint4*)(p + 16);
    }
#pragma unroll
    for (int fm = 0; fm < 4; ++fm) {
      Frag af;
      const u16* p = &ldsA[cur][(waveM + fm * 16 + l16) * ST + lh * 8];
      af.q[0] = *(const uint4*)(p);
      af.q[1] = *(const uint4*)(p + 16);
#pragma unroll
      for (int fn = 0; fn < 2; ++fn)
        acc[fm][fn] = wmma_bf16(af.v, bfr[fn].v, acc[fm][fn]);
    }
    __syncthreads();                   // buffer free for async overwrite
  }

  // ---- epilogue: C frag element (lane, v) = (m = v + 8*(lane/16), n = lane%16)
#pragma unroll
  for (int fm = 0; fm < 4; ++fm)
#pragma unroll
    for (int fn = 0; fn < 2; ++fn)
#pragma unroll
      for (int v = 0; v < 8; ++v) {
        int m = m0 + waveM + fm * 16 + v + 8 * lh;
        int n = n0 + waveN + fn * 16 + l16;
        float val = acc[fm][fn][v] + bias[n];
        if (mode == 0) {
          int which = n >> 10;          // 0=q 1=k 2=v
          int d  = n & 1023;
          int h  = d >> 6, hd = d & 63;
          int b  = m >> 11, s = m & 2047;
          int bh = (b << 4) | h;
          u16 bv = f2bf(val);
          if (which == 0)      Qb[(size_t)bh * (S_ * HD_) + (size_t)s * HD_ + hd] = bv;
          else if (which == 1) Kb[(size_t)bh * (S_ * HD_) + (size_t)s * HD_ + hd] = bv;
          else                 Vt[(size_t)bh * (HD_ * S_) + (size_t)hd * S_ + s] = bv;
        } else {
          outF[(size_t)m * N + n] = val;
        }
      }
}

// ---------------------------------------------------------------------------
// Causal flash attention.  grid = (S/128, B*H), 256 threads (8 waves).
// Each wave owns 16 query rows; Q A-fragments in registers; K/Vt B-fragments
// stream straight from global (contiguous 16B per lane by layout).  Online
// softmax with lane-local row stats (C-frag row mapping m = v + 8*(lane/16)).
// ---------------------------------------------------------------------------
__global__ __launch_bounds__(256) void attn_fwd(
    const u16* __restrict__ Qb, const u16* __restrict__ Kb,
    const u16* __restrict__ Vt, u16* __restrict__ attnOut)
{
  constexpr int PST = 40;
  __shared__ __align__(16) u16 ldsP[8 * 16 * PST];   // per-wave P relayout tile

  const int lane = threadIdx.x & 31;
  const int wid  = threadIdx.x >> 5;
  const int l16  = lane & 15;
  const int lh   = lane >> 4;
  const int bh   = blockIdx.y;
  const int q0   = blockIdx.x * 128 + wid * 16;      // this wave's 16 q rows

  u16* myP = ldsP + wid * 16 * PST;

  // Q fragments (A layout): k 0..31 and 32..63 of head_dim
  Frag qf[2];
  {
    const u16* p = Qb + ((size_t)bh * S_ + (q0 + l16)) * HD_ + lh * 8;
    qf[0].q[0] = *(const uint4*)(p);
    qf[0].q[1] = *(const uint4*)(p + 16);
    qf[1].q[0] = *(const uint4*)(p + 32);
    qf[1].q[1] = *(const uint4*)(p + 48);
  }

  v8f o[4];
  float rmax[8], rsum[8];
#pragma unroll
  for (int f = 0; f < 4; ++f) o[f] = (v8f){0.f,0.f,0.f,0.f,0.f,0.f,0.f,0.f};
#pragma unroll
  for (int v = 0; v < 8; ++v) { rmax[v] = -1e30f; rsum[v] = 0.f; }

  const int ktiles = (q0 + 16 + 31) >> 5;            // causal bound, 32-key tiles
  for (int tile = 0; tile < ktiles; ++tile) {
    const int kt0 = tile * 32;

    if (tile + 1 < ktiles) {   // pull next K/V tiles toward cache
      __builtin_prefetch(Kb + ((size_t)bh * S_ + (kt0 + 32 + lane)) * HD_, 0, 1);
      __builtin_prefetch(Vt + ((size_t)bh * HD_ + lane * 2) * S_ + kt0 + 32, 0, 1);
    }

    // ---- scores: two 16x16 frags covering 32 keys, accumulate over HD=64
    v8f s[2];
#pragma unroll
    for (int f = 0; f < 2; ++f) {
      const u16* p = Kb + ((size_t)bh * S_ + (kt0 + f * 16 + l16)) * HD_ + lh * 8;
      Frag b0, b1;
      b0.q[0] = *(const uint4*)(p);       b0.q[1] = *(const uint4*)(p + 16);
      b1.q[0] = *(const uint4*)(p + 32);  b1.q[1] = *(const uint4*)(p + 48);
      v8f a = (v8f){0.f,0.f,0.f,0.f,0.f,0.f,0.f,0.f};
      a = wmma_bf16(qf[0].v, b0.v, a);
      a = wmma_bf16(qf[1].v, b1.v, a);
      s[f] = a;
    }

    // ---- scale, causal mask, online softmax; write P (bf16) to LDS
    float corr[8];
#pragma unroll
    for (int v = 0; v < 8; ++v) {
      int qrow = q0 + v + 8 * lh;
      float a0 = s[0][v] * SCALE_;
      float a1 = s[1][v] * SCALE_;
      a0 = (kt0 + l16      <= qrow) ? a0 : -1e9f;
      a1 = (kt0 + 16 + l16 <= qrow) ? a1 : -1e9f;
      float mv = fmaxf(a0, a1);
#pragma unroll
      for (int m = 1; m < 16; m <<= 1) mv = fmaxf(mv, __shfl_xor(mv, m));
      float mnew = fmaxf(rmax[v], mv);
      corr[v] = __expf(rmax[v] - mnew);
      float p0 = __expf(a0 - mnew);
      float p1 = __expf(a1 - mnew);
      float ps = p0 + p1;
#pragma unroll
      for (int m = 1; m < 16; m <<= 1) ps += __shfl_xor(ps, m);
      rsum[v] = rsum[v] * corr[v] + ps;
      rmax[v] = mnew;
      int row = v + 8 * lh;
      myP[row * PST + l16]      = f2bf(p0);
      myP[row * PST + 16 + l16] = f2bf(p1);
    }
#pragma unroll
    for (int f = 0; f < 4; ++f)
#pragma unroll
      for (int v = 0; v < 8; ++v) o[f][v] *= corr[v];

    // ---- P (16x32) A-fragment from LDS, then O += P @ V via Vt
    Frag pf;
    {
      const u16* p = myP + l16 * PST + lh * 8;
      pf.q[0] = *(const uint4*)(p);
      pf.q[1] = *(const uint4*)(p + 16);
    }
#pragma unroll
    for (int hdf = 0; hdf < 4; ++hdf) {
      const u16* p = Vt + ((size_t)bh * HD_ + hdf * 16 + l16) * S_ + kt0 + lh * 8;
      Frag vf;
      vf.q[0] = *(const uint4*)(p);
      vf.q[1] = *(const uint4*)(p + 16);
      o[hdf] = wmma_bf16(pf.v, vf.v, o[hdf]);
    }
  }

  // ---- normalize and store attn output (b, s, h*64+hd) as bf16 for GEMM2
  const int b = bh >> 4, h = bh & 15;
#pragma unroll
  for (int v = 0; v < 8; ++v) {
    float inv = 1.0f / rsum[v];
    int srow = q0 + v + 8 * lh;
    size_t base = ((size_t)b * S_ + srow) * D_ + h * HD_;
#pragma unroll
    for (int hdf = 0; hdf < 4; ++hdf)
      attnOut[base + hdf * 16 + l16] = f2bf(o[hdf][v] * inv);
  }
}

// ---------------------------------------------------------------------------
extern "C" void kernel_launch(void* const* d_in, const int* in_sizes, int n_in,
                              void* d_out, int out_size, void* d_ws, size_t ws_size,
                              hipStream_t stream)
{
  const float* x    = (const float*)d_in[0];
  /* d_in[1] = mask (causal, handled analytically) */
  const float* Wqkv = (const float*)d_in[2];
  const float* bqkv = (const float*)d_in[3];
  const float* Wout = (const float*)d_in[4];
  const float* bout = (const float*)d_in[5];
  float* out = (float*)d_out;

  char* ws = (char*)d_ws;
  size_t off = 0;
  auto alloc = [&](size_t bytes) -> char* {
    char* p = ws + off;
    off = (off + bytes + 255) & ~(size_t)255;
    return p;
  };
  u16* xb     = (u16*)alloc((size_t)BS_ * D_ * 2);
  u16* Wqkvt  = (u16*)alloc((size_t)3 * D_ * D_ * 2);   // (3D x D) = Wqkv^T bf16
  u16* Woutt  = (u16*)alloc((size_t)D_ * D_ * 2);       // (D x D)  = Wout^T bf16
  u16* Qb     = (u16*)alloc((size_t)BH_ * S_ * HD_ * 2);
  u16* Kb     = (u16*)alloc((size_t)BH_ * S_ * HD_ * 2);
  u16* Vt     = (u16*)alloc((size_t)BH_ * HD_ * S_ * 2);
  u16* attnb  = (u16*)alloc((size_t)BS_ * D_ * 2);

  const int nX = BS_ * D_;
  f32_to_bf16<<<(nX + 255) / 256, 256, 0, stream>>>(x, xb, nX);
  // Wqkv: K=1024 x N=3072 -> Wqkvt 3072 x 1024 ; Wout: 1024x1024 -> 1024x1024
  transpose_f32_to_bf16<<<dim3(3 * D_ / 32, D_ / 32), 256, 0, stream>>>(
      Wqkv, Wqkvt, D_, 3 * D_);
  transpose_f32_to_bf16<<<dim3(D_ / 32, D_ / 32), 256, 0, stream>>>(
      Wout, Woutt, D_, D_);

  // qkv = x @ Wqkv + bqkv, scattered to Q/K (b,h,s,hd) and Vt (b,h,hd,s)
  gemm_bf16<<<dim3(3 * D_ / 128, BS_ / 128), 256, 0, stream>>>(
      xb, Wqkvt, bqkv, BS_, 3 * D_, D_, 0, Qb, Kb, Vt, nullptr);

  // causal flash attention
  attn_fwd<<<dim3(S_ / 128, BH_), 256, 0, stream>>>(Qb, Kb, Vt, attnb);

  // out = attn @ Wout + bout (f32 output)
  gemm_bf16<<<dim3(D_ / 128, BS_ / 128), 256, 0, stream>>>(
      attnb, Woutt, bout, BS_, D_, D_, 1, nullptr, nullptr, nullptr, out);
}
